// Euclidean_26336739459570
// MI455X (gfx1250) — matrix operand order repaired and checked by reading, not compiled, not run on hardware
//
#include <hip/hip_runtime.h>

#define IN_F      2048
#define N_CLASSES 1000
#define B_ROWS    16384

#define BM 128
#define BN 256
#define BK 32
#define LDT 40                 // padded LDS row stride in bf16 elems (80B = 20 banks -> conflict free)
#define KTILES (IN_F / BK)     // 64

typedef __attribute__((ext_vector_type(4)))  float   f32x4;
typedef __attribute__((ext_vector_type(8)))  float   v8f;
typedef __attribute__((ext_vector_type(4)))  __bf16  bf16x4;
typedef __attribute__((ext_vector_type(8)))  __bf16  bf16x8;
typedef __attribute__((ext_vector_type(16))) __bf16  v16bf;

// ---------------- exact f32 row sum-of-squares (one wave per row) ----------------
__global__ __launch_bounds__(256)
void rowsq_kernel(const float* __restrict__ src, float* __restrict__ dst, int nrows) {
    const int wave = threadIdx.x >> 5;
    const int lane = threadIdx.x & 31;
    const int row  = blockIdx.x * 8 + wave;
    if (row >= nrows) return;
    const f32x4* p = (const f32x4*)(src + (size_t)row * IN_F);
    float s = 0.0f;
#pragma unroll
    for (int i = 0; i < IN_F / 4 / 32; ++i) {
        f32x4 v = p[i * 32 + lane];
        s += v.x * v.x + v.y * v.y + v.z * v.z + v.w * v.w;
    }
#pragma unroll
    for (int off = 16; off > 0; off >>= 1) s += __shfl_down(s, off, 32);
    if (lane == 0) dst[row] = s;
}

// ---------------- bf16 WMMA GEMM with fused distance epilogue ----------------
// Block tile: 128(M) x 256(N), K staged in 32-slices, double-buffered LDS.
// 8 waves, each owning a 64x64 sub-tile (4x4 WMMA accumulator tiles).
__global__ __launch_bounds__(256)
void euclid_gemm_kernel(const float* __restrict__ X, const float* __restrict__ W,
                        const float* __restrict__ x2, const float* __restrict__ w2,
                        float* __restrict__ Out) {
    __shared__ __bf16 As[2][BM * LDT];
    __shared__ __bf16 Bs[2][BN * LDT];

    const int tid  = threadIdx.x;
    const int lane = tid & 31;
    const int wave = tid >> 5;
    const int wm   = wave & 1;   // 0..1 : 64 rows each
    const int wn   = wave >> 1;  // 0..3 : 64 cols each

    const int r0 = blockIdx.y * BM;
    const int c0 = blockIdx.x * BN;

    // global load mapping: 16B per thread per row-group; 4 groups for X, 8 for W
    const int grow = tid >> 3;   // 0..31 (rows, stepped by 32 per group)
    const int gc4  = tid & 7;    // 0..7  (float4 column within BK=32)

    f32x4 xr[4], wr[8];
    auto g_load = [&](int kt) {
        const int k0 = kt * BK;
#pragma unroll
        for (int i = 0; i < 4; ++i) {
            const int row = i * 32 + grow;
            xr[i] = *(const f32x4*)(X + (size_t)(r0 + row) * IN_F + k0 + gc4 * 4);
        }
#pragma unroll
        for (int i = 0; i < 8; ++i) {
            const int row = i * 32 + grow;
            int wrow = c0 + row;
            wrow = wrow < N_CLASSES ? wrow : (N_CLASSES - 1);   // clamp padded cols
            wr[i] = *(const f32x4*)(W + (size_t)wrow * IN_F + k0 + gc4 * 4);
        }
    };
    auto s_store = [&](int buf) {
#pragma unroll
        for (int i = 0; i < 4; ++i) {
            const int row = i * 32 + grow;
            *(bf16x4*)(&As[buf][row * LDT + gc4 * 4]) = __builtin_convertvector(xr[i], bf16x4);
        }
#pragma unroll
        for (int i = 0; i < 8; ++i) {
            const int row = i * 32 + grow;
            *(bf16x4*)(&Bs[buf][row * LDT + gc4 * 4]) = __builtin_convertvector(wr[i], bf16x4);
        }
    };

    // WMMA fragment addressing (ISA 7.12.2, 16-bit A 16x32 / B 32x16 layouts)
    const int am  = lane & 15;           // A row within tile
    const int akb = (lane >> 4) << 3;    // A K-chunk base: 0 or 8
    const int bn  = lane & 15;           // B col within tile
    const int bkh = (lane >> 4) << 3;    // B K-chunk base: 0 or 8

    v8f vzero = {};
    v8f acc[4][4];
#pragma unroll
    for (int mt = 0; mt < 4; ++mt)
#pragma unroll
        for (int nt = 0; nt < 4; ++nt) acc[mt][nt] = vzero;

    g_load(0);
    s_store(0);
    __syncthreads();

    for (int kt = 0; kt < KTILES; ++kt) {
        const int cur = kt & 1;
        if (kt + 1 < KTILES) g_load(kt + 1);   // prefetch next slice while computing

        v16bf afr[4];
#pragma unroll
        for (int mt = 0; mt < 4; ++mt) {
            const __bf16* base = &As[cur][(wm * 64 + mt * 16 + am) * LDT + akb];
            bf16x8 lo = *(const bf16x8*)(base);
            bf16x8 hi = *(const bf16x8*)(base + 16);
            afr[mt] = __builtin_shufflevector(lo, hi, 0, 1, 2, 3, 4, 5, 6, 7,
                                                       8, 9, 10, 11, 12, 13, 14, 15);
        }
        v16bf bfr[4];
#pragma unroll
        for (int nt = 0; nt < 4; ++nt) {
            const __bf16* base = &Bs[cur][(wn * 64 + nt * 16 + bn) * LDT + bkh];
            bf16x8 lo = *(const bf16x8*)(base);
            bf16x8 hi = *(const bf16x8*)(base + 16);
            bfr[nt] = __builtin_shufflevector(lo, hi, 0, 1, 2, 3, 4, 5, 6, 7,
                                                       8, 9, 10, 11, 12, 13, 14, 15);
        }

#pragma unroll
        for (int mt = 0; mt < 4; ++mt)
#pragma unroll
            for (int nt = 0; nt < 4; ++nt)
                acc[mt][nt] = __builtin_amdgcn_wmma_f32_16x16x32_bf16(
                    false, afr[mt], false, bfr[nt], (short)0, acc[mt][nt], false, false);

        if (kt + 1 < KTILES) s_store(cur ^ 1);
        __syncthreads();
    }

    // epilogue: out = (2*xw - x2 - w2) / F
    const float inv = 1.0f / (float)IN_F;
    const int mlo = (lane >> 4) * 8;   // C/D layout: m = r + 8*(lane>=16)
    const int cn  = lane & 15;         // n = lane & 15

    float w2v[4];
    int   colv[4];
#pragma unroll
    for (int nt = 0; nt < 4; ++nt) {
        colv[nt] = c0 + wn * 64 + nt * 16 + cn;
        w2v[nt]  = w2[colv[nt] < N_CLASSES ? colv[nt] : (N_CLASSES - 1)];
    }

#pragma unroll
    for (int mt = 0; mt < 4; ++mt) {
        const int rowb = r0 + wm * 64 + mt * 16 + mlo;
        float xx[8];
#pragma unroll
        for (int r = 0; r < 8; ++r) xx[r] = x2[rowb + r];
#pragma unroll
        for (int nt = 0; nt < 4; ++nt) {
            if (colv[nt] < N_CLASSES) {
#pragma unroll
                for (int r = 0; r < 8; ++r) {
                    const float v = (2.0f * acc[mt][nt][r] - xx[r] - w2v[nt]) * inv;
                    Out[(size_t)(rowb + r) * N_CLASSES + colv[nt]] = v;
                }
            }
        }
    }
}

extern "C" void kernel_launch(void* const* d_in, const int* in_sizes, int n_in,
                              void* d_out, int out_size, void* d_ws, size_t ws_size,
                              hipStream_t stream) {
    (void)in_sizes; (void)n_in; (void)out_size; (void)ws_size;
    const float* x = (const float*)d_in[0];   // [16384, 2048] f32
    const float* w = (const float*)d_in[1];   // [1000, 2048] f32
    float* out = (float*)d_out;               // [16384, 1000] f32

    float* x2 = (float*)d_ws;                 // 16384 f32
    float* w2 = x2 + B_ROWS;                  // 1000  f32

    rowsq_kernel<<<B_ROWS / 8, 256, 0, stream>>>(x, x2, B_ROWS);
    rowsq_kernel<<<(N_CLASSES + 7) / 8, 256, 0, stream>>>(w, w2, N_CLASSES);

    dim3 grid((N_CLASSES + BN - 1) / BN, B_ROWS / BM);  // 4 x 128
    euclid_gemm_kernel<<<grid, 256, 0, stream>>>(x, w, x2, w2, out);
}
